// MultiheadAttention_13881334301006
// MI455X (gfx1250) — matrix operand order repaired and checked by reading, not compiled, not run on hardware
//
#include <hip/hip_runtime.h>
#include <stdint.h>

#define HIDDEN 768
#define NHEADS 12
#define HEADD  64
#define BATCH  32
#define SEQ    512

typedef __attribute__((ext_vector_type(16))) _Float16 v16h;
typedef __attribute__((ext_vector_type(8)))  float    v8f;

union Frag { v16h v; uint4 u[2]; };

// ---------------- fp32 -> f16 conversion ----------------
__global__ void cvt_f32_f16(const float* __restrict__ in,
                            _Float16* __restrict__ out, size_t n) {
  size_t i = (size_t)blockIdx.x * blockDim.x + threadIdx.x;
  if (i < n) out[i] = (_Float16)in[i];
}

// ------------- fp32 [K][N] -> f16 [N][K] transpose (weights, once) -------------
__global__ __launch_bounds__(256)
void cvt_transpose_f32_f16(const float* __restrict__ in,
                           _Float16* __restrict__ out, int K, int N) {
  __shared__ float t[32][33];
  const int n0 = blockIdx.x * 32;
  const int k0 = blockIdx.y * 32;
  const int c  = threadIdx.x & 31;
  const int r0 = threadIdx.x >> 5;  // 0..7
#pragma unroll
  for (int i = 0; i < 4; ++i) {
    int r = r0 + i * 8;
    t[r][c] = in[(size_t)(k0 + r) * N + n0 + c];
  }
  __syncthreads();
#pragma unroll
  for (int i = 0; i < 4; ++i) {
    int r = r0 + i * 8;
    out[(size_t)(n0 + r) * K + k0 + c] = (_Float16)t[c][r];
  }
}

// ---------------- tiled WMMA GEMM: C[M,N] = A[M,K]*Bt[N,K]^T + bias ----------------
// 256 threads (8 waves), block tile 256x64, K-step 32. Wave owns 32x64 (8 accums).
// Bt is the PRE-TRANSPOSED weight [N][K] so all staging is straight b128 copies.
// MODE: 0 = f16 [M][N] out, 1 = f32 [M][N] out, 2 = f16 out transposed per (b,h):
//       Vt[((b*NHEADS+h)*HEADD+d)*SEQ + s]  (for direct V B-frag reads in attention)
template<int MODE>
__global__ __launch_bounds__(256)
void gemm_f16_wmma(const _Float16* __restrict__ A,
                   const _Float16* __restrict__ Bt,
                   const float* __restrict__ bias,
                   void* __restrict__ C,
                   int M, int N, int K) {
  __shared__ __align__(16) _Float16 As[256][40];  // [m][k], 80B rows (16B mult)
  __shared__ __align__(16) _Float16 Bs[64][40];   // [n][k]

  const int tid  = threadIdx.x;
  const int lane = tid & 31;
  const int wv   = tid >> 5;
  const int ln   = lane & 15;
  const int hg   = lane >> 4;
  const int m0   = blockIdx.x * 256;
  const int n0   = blockIdx.y * 64;

  v8f zero = {0.f,0.f,0.f,0.f,0.f,0.f,0.f,0.f};
  v8f acc[2][4];
#pragma unroll
  for (int s = 0; s < 2; ++s)
#pragma unroll
    for (int t = 0; t < 4; ++t) acc[s][t] = zero;

  // register double-buffer for global->LDS staging
  uint4 pa[4], pb;
  const int bn_ld = tid >> 2;          // 0..63
  const int bk_ld = (tid & 3) * 8;     // 0,8,16,24
  auto loadTiles = [&](int kk) {
    const uint4* sa = (const uint4*)(A + (size_t)(m0 + tid) * K + kk);
    pa[0] = sa[0]; pa[1] = sa[1]; pa[2] = sa[2]; pa[3] = sa[3];
    pb = *(const uint4*)(Bt + (size_t)(n0 + bn_ld) * K + kk + bk_ld);
  };
  auto storeTiles = [&]() {
    *(uint4*)&As[tid][0]  = pa[0];
    *(uint4*)&As[tid][8]  = pa[1];
    *(uint4*)&As[tid][16] = pa[2];
    *(uint4*)&As[tid][24] = pa[3];
    *(uint4*)&Bs[bn_ld][bk_ld] = pb;
  };

  loadTiles(0);
  storeTiles();
  __syncthreads();

  for (int kk = 0; kk < K; kk += 32) {
    const bool more = (kk + 32) < K;
    if (more) loadTiles(kk + 32);  // overlap next-tile global loads with WMMA

    Frag af[2];
#pragma unroll
    for (int s = 0; s < 2; ++s) {
      int ar = wv * 32 + s * 16 + ln;
      int k0 = hg * 8;
      af[s].u[0] = *(const uint4*)&As[ar][k0];
      af[s].u[1] = *(const uint4*)&As[ar][16 + k0];
    }
#pragma unroll
    for (int t = 0; t < 4; ++t) {
      Frag bf;  // B frag: lane holds col n=ln, K rows (hg*8..+7, 16+hg*8..+7)
      bf.u[0] = *(const uint4*)&Bs[t * 16 + ln][hg * 8];
      bf.u[1] = *(const uint4*)&Bs[t * 16 + ln][16 + hg * 8];
#pragma unroll
      for (int s = 0; s < 2; ++s)
        acc[s][t] = __builtin_amdgcn_wmma_f32_16x16x32_f16(
            false, af[s].v, false, bf.v, (short)0, acc[s][t], false, false);
    }
    __syncthreads();
    if (more) { storeTiles(); __syncthreads(); }
  }

  // epilogue: C layout -> lane ln is column, VGPR j is row j+8*hg
#pragma unroll
  for (int s = 0; s < 2; ++s)
#pragma unroll
    for (int t = 0; t < 4; ++t) {
      int ncol = n0 + t * 16 + ln;
      float bv = bias[ncol];
#pragma unroll
      for (int j = 0; j < 8; ++j) {
        int mr = m0 + wv * 32 + s * 16 + hg * 8 + j;
        float val = acc[s][t][j] + bv;
        if (MODE == 1) {
          ((float*)C)[(size_t)mr * N + ncol] = val;
        } else if (MODE == 0) {
          ((_Float16*)C)[(size_t)mr * N + ncol] = (_Float16)val;
        } else {
          int bb = mr >> 9, ss = mr & (SEQ - 1);
          int hh = ncol >> 6, dd = ncol & (HEADD - 1);
          ((_Float16*)C)[(((size_t)(bb * NHEADS + hh) * HEADD + dd) << 9) + ss] =
              (_Float16)val;
        }
      }
    }
}

// ---------------- flash attention ----------------
// grid: (SEQ/64, BATCH*NHEADS), 128 threads (4 waves). Wave owns 16 query rows.
// Q/K read directly from [B,S,HIDDEN] f16 (d-contiguous); V read from Vt (key-
// contiguous). No block-level sync at all; only a wave-private P bounce in LDS.
__global__ __launch_bounds__(128)
void flash_attn(const _Float16* __restrict__ Q,
                const _Float16* __restrict__ Km,
                const _Float16* __restrict__ Vt,
                const float* __restrict__ mask,
                _Float16* __restrict__ O) {
  __shared__ __align__(16) _Float16 Ps[4][16][72];  // per-wave P tile [m][key]

  const int tid  = threadIdx.x;
  const int lane = tid & 31;
  const int wv   = tid >> 5;
  const int ln   = lane & 15;
  const int hg   = lane >> 4;
  const int qb   = blockIdx.x * 64;
  const int b    = blockIdx.y / NHEADS;
  const int h    = blockIdx.y % NHEADS;

  const _Float16* vbase = Vt + (size_t)(b * NHEADS + h) * HEADD * SEQ;

  // Q A-frags, HEAD=64 -> two K=32 steps
  Frag qa[2];
  {
    const _Float16* qp =
        Q + (size_t)(b * SEQ + qb + wv * 16 + ln) * HIDDEN + h * HEADD;
#pragma unroll
    for (int s = 0; s < 2; ++s) {
      qa[s].u[0] = *(const uint4*)(qp + s * 32 + hg * 8);
      qa[s].u[1] = *(const uint4*)(qp + s * 32 + 16 + hg * 8);
    }
  }

  v8f zero = {0.f,0.f,0.f,0.f,0.f,0.f,0.f,0.f};
  v8f oacc[4];
#pragma unroll
  for (int t = 0; t < 4; ++t) oacc[t] = zero;
  float rmax[8], rsum[8];
#pragma unroll
  for (int j = 0; j < 8; ++j) { rmax[j] = -3.0e38f; rsum[j] = 0.f; }

  for (int kb2 = 0; kb2 < SEQ / 64; ++kb2) {
    // prefetch next key-block of K and Vt into cache (global_prefetch_b8)
    if (kb2 + 1 < SEQ / 64) {
      int nb = kb2 + 1;
      __builtin_prefetch(Km + (size_t)(b * SEQ + nb * 64 + (tid >> 1)) * HIDDEN +
                             h * HEADD + (tid & 1) * 32, 0, 3);
      __builtin_prefetch(vbase + (size_t)(tid >> 1) * SEQ + nb * 64 +
                             (tid & 1) * 32, 0, 3);
    }

    // scores S = Q*K^T / 8 + mask ; K^T B-frag is d-contiguous in global
    float sc[4][8];
#pragma unroll
    for (int t = 0; t < 4; ++t) {
      v8f sa = zero;
#pragma unroll
      for (int s = 0; s < 2; ++s) {
        Frag kf;
        const _Float16* kp =
            Km + (size_t)(b * SEQ + kb2 * 64 + t * 16 + ln) * HIDDEN +
            h * HEADD + s * 32 + hg * 8;
        kf.u[0] = *(const uint4*)(kp);
        kf.u[1] = *(const uint4*)(kp + 16);
        sa = __builtin_amdgcn_wmma_f32_16x16x32_f16(
            false, qa[s].v, false, kf.v, (short)0, sa, false, false);
      }
      float mv = mask[(size_t)b * SEQ + kb2 * 64 + t * 16 + ln];
#pragma unroll
      for (int j = 0; j < 8; ++j) sc[t][j] = sa[j] * 0.125f + mv;
    }

    // online softmax per row (row = j + 8*hg; reduce across 16 lanes of half)
#pragma unroll
    for (int j = 0; j < 8; ++j) {
      float mx = fmaxf(fmaxf(sc[0][j], sc[1][j]), fmaxf(sc[2][j], sc[3][j]));
      mx = fmaxf(mx, __shfl_xor(mx, 1, 32));
      mx = fmaxf(mx, __shfl_xor(mx, 2, 32));
      mx = fmaxf(mx, __shfl_xor(mx, 4, 32));
      mx = fmaxf(mx, __shfl_xor(mx, 8, 32));
      float mnew = fmaxf(rmax[j], mx);
      float corr = __expf(rmax[j] - mnew);
      rmax[j] = mnew;
      float ls = 0.f;
#pragma unroll
      for (int t = 0; t < 4; ++t) {
        float p = __expf(sc[t][j] - mnew);
        sc[t][j] = p;
        ls += p;
      }
      ls += __shfl_xor(ls, 1, 32);
      ls += __shfl_xor(ls, 2, 32);
      ls += __shfl_xor(ls, 4, 32);
      ls += __shfl_xor(ls, 8, 32);
      rsum[j] = rsum[j] * corr + ls;
#pragma unroll
      for (int t = 0; t < 4; ++t) oacc[t][j] *= corr;
    }

    // bounce P through wave-private LDS to convert C-layout -> A-layout
#pragma unroll
    for (int t = 0; t < 4; ++t)
#pragma unroll
      for (int j = 0; j < 8; ++j)
        Ps[wv][hg * 8 + j][t * 16 + ln] = (_Float16)sc[t][j];
    __builtin_amdgcn_wave_barrier();  // pin DS write->read order (DS in-order per wave)

    // O += P * V  (V B-frag: lane holds col d=t*16+ln, keys contiguous in Vt)
#pragma unroll
    for (int s = 0; s < 2; ++s) {
      Frag pa;
      pa.u[0] = *(const uint4*)&Ps[wv][ln][s * 32 + hg * 8];
      pa.u[1] = *(const uint4*)&Ps[wv][ln][s * 32 + 16 + hg * 8];
#pragma unroll
      for (int t = 0; t < 4; ++t) {
        Frag vf;
        const _Float16* vp =
            vbase + (size_t)(t * 16 + ln) * SEQ + kb2 * 64 + s * 32 + hg * 8;
        vf.u[0] = *(const uint4*)(vp);
        vf.u[1] = *(const uint4*)(vp + 16);
        oacc[t] = __builtin_amdgcn_wmma_f32_16x16x32_f16(
            false, pa.v, false, vf.v, (short)0, oacc[t], false, false);
      }
    }
  }

  // normalize and store attn (f16) back to [B,S,HIDDEN] layout
#pragma unroll
  for (int t = 0; t < 4; ++t)
#pragma unroll
    for (int j = 0; j < 8; ++j) {
      float val = oacc[t][j] / rsum[j];
      int row = b * SEQ + qb + wv * 16 + hg * 8 + j;
      int col = h * HEADD + t * 16 + ln;
      O[(size_t)row * HIDDEN + col] = (_Float16)val;
    }
}

// ---------------- host side ----------------
extern "C" void kernel_launch(void* const* d_in, const int* in_sizes, int n_in,
                              void* d_out, int out_size, void* d_ws, size_t ws_size,
                              hipStream_t stream) {
  const float* x    = (const float*)d_in[0];
  const float* mask = (const float*)d_in[1];
  const float* Wq   = (const float*)d_in[2];
  const float* bq   = (const float*)d_in[3];
  const float* Wk   = (const float*)d_in[4];
  const float* bk   = (const float*)d_in[5];
  const float* Wv   = (const float*)d_in[6];
  const float* bv   = (const float*)d_in[7];
  const float* Wo   = (const float*)d_in[8];
  const float* bo   = (const float*)d_in[9];
  float* out = (float*)d_out;

  const size_t MTOT = (size_t)BATCH * SEQ;       // 16384
  const size_t XSZ  = MTOT * HIDDEN;             // 12.58M elems
  const size_t WSZ  = (size_t)HIDDEN * HIDDEN;

  _Float16* p   = (_Float16*)d_ws;
  _Float16* xh  = p; p += XSZ;
  _Float16* wqt = p; p += WSZ;   // transposed weights [N][K]
  _Float16* wkt = p; p += WSZ;
  _Float16* wvt = p; p += WSZ;
  _Float16* wot = p; p += WSZ;
  _Float16* qh  = p; p += XSZ;
  _Float16* kh  = p; p += XSZ;
  _Float16* vth = p; p += XSZ;   // V transposed per (b,h): [b][h][d][s]
  _Float16* ah  = p; p += XSZ;

  const int thr = 256;
  cvt_f32_f16<<<(unsigned)((XSZ + thr - 1) / thr), thr, 0, stream>>>(x, xh, XSZ);

  dim3 gt(HIDDEN / 32, HIDDEN / 32);
  cvt_transpose_f32_f16<<<gt, 256, 0, stream>>>(Wq, wqt, HIDDEN, HIDDEN);
  cvt_transpose_f32_f16<<<gt, 256, 0, stream>>>(Wk, wkt, HIDDEN, HIDDEN);
  cvt_transpose_f32_f16<<<gt, 256, 0, stream>>>(Wv, wvt, HIDDEN, HIDDEN);
  cvt_transpose_f32_f16<<<gt, 256, 0, stream>>>(Wo, wot, HIDDEN, HIDDEN);

  dim3 gg((unsigned)(MTOT / 256), HIDDEN / 64);
  gemm_f16_wmma<0><<<gg, 256, 0, stream>>>(xh, wqt, bq, qh,  (int)MTOT, HIDDEN, HIDDEN);
  gemm_f16_wmma<0><<<gg, 256, 0, stream>>>(xh, wkt, bk, kh,  (int)MTOT, HIDDEN, HIDDEN);
  gemm_f16_wmma<2><<<gg, 256, 0, stream>>>(xh, wvt, bv, vth, (int)MTOT, HIDDEN, HIDDEN);

  dim3 ga(SEQ / 64, BATCH * NHEADS);
  flash_attn<<<ga, 128, 0, stream>>>(qh, kh, vth, mask, ah);

  gemm_f16_wmma<1><<<gg, 256, 0, stream>>>(ah, wot, bo, out, (int)MTOT, HIDDEN, HIDDEN);
}
